// PositionalEncoding_78494822302098
// MI455X (gfx1250) — compile-verified, matrix operand order checked by reading, not run on hardware
//
#include <hip/hip_runtime.h>

#define DEV __device__ __forceinline__

typedef __attribute__((ext_vector_type(16))) _Float16     v16h;
typedef __attribute__((ext_vector_type(8)))  float        v8f;
typedef __attribute__((ext_vector_type(4)))  unsigned int v4ui;
typedef __attribute__((ext_vector_type(8)))  int          v8i;
typedef __attribute__((ext_vector_type(4)))  int          v4i;

#if __has_builtin(__builtin_amdgcn_tensor_load_to_lds)
#define HAVE_TDM 1
#else
#define HAVE_TDM 0
#endif

// ---- problem constants --------------------------------------------------
#define B_     256
#define N_     1024
#define K_     64
#define E_     128
#define ROWS   64           // output rows per workgroup
#define TPB    256          // 8 wave32 per workgroup
#define LN_EPS 1e-5f
#define TWO_PI 6.28318530717958647692f

// ---- LDS pool with region reuse across barriers -------------------------
// All WMMA operands are stored fragment-major ("swizzled"): 1 KB panels of
// 32 lanes x 16 halves, so a v16h fragment load is 2x ds_load_b128.
#define OFF_COSA   0u        // 8 KB f16 A-type 64x64   } reused as xA
#define OFF_SINA   8192u     // 8 KB f16 A-type 64x64   }
#define OFF_XA     0u        // 16 KB f16 A-type 64x128
#define OFF_ADJ    16384u    // 8 KB f16 B-type 64x64   } reused by W2 (32KB)
#define OFF_WR     24576u    // 8 KB f16 A-type (weighted real)
#define OFF_WI     32768u    // 8 KB f16 A-type (weighted imag)
#define OFF_WRB    40960u    // 8 KB f16 B-type (w_real[sg])
#define OFF_W2     16384u    // 32 KB f16 B-type 128x128 (W2^T)
#define OFF_WIB    49152u    // 8 KB f16 B-type (w_imag[sg]) } reused by xOut
#define OFF_XOUT   49152u    // 32 KB f32 (64x128 pre-LN, row-major)
#define OFF_TABC   81920u    // 192 f32
#define OFF_POST   82688u    // 192 f32
#define OFF_BR     83456u    // 64 f32
#define OFF_BI     83712u    // 64 f32
#define OFF_B2     83968u    // 128 f32
#define OFF_GAM    84480u    // 128 f32
#define OFF_BET    84992u    // 128 f32
#define OFF_MU     85504u    // 64 f32
#define OFF_RS     85760u    // 64 f32
#if HAVE_TDM
#define OFF_W2STG  86016u    // 64 KB f32 TDM staging for W2 (row-major)
#define SMEM_BYTES (86016u + 65536u)
#else
#define SMEM_BYTES 86016u
#endif

// ---- WMMA + fragment helpers (ISA 7.12.2 layouts, wave32) ---------------
DEV v8f wmma_f16(v16h a, v16h b, v8f c) {
  return __builtin_amdgcn_wmma_f32_16x16x32_f16(
      /*neg_a=*/false, a, /*neg_b=*/false, b,
      /*c_mod=*/(short)0, c, /*reuse_a=*/false, /*reuse_b=*/false);
}

// Swizzled element index for an A-operand (M x K), kPanels = K/32.
// lane = (m&15) + 16*bit3(k&31); element j = ((k&31)>>4)*8 | (k&7).
DEV int idxA(int m, int k, int kPanels) {
  const int kl = k & 31;
  const int panel = (m >> 4) * kPanels + (k >> 5);
  const int lane = (m & 15) + ((kl & 8) << 1);
  const int j = ((kl >> 4) << 3) | (kl & 7);
  return panel * 512 + lane * 16 + j;
}
// Swizzled element index for a B-operand (K x N), nPanels = N/16.
DEV int idxB(int k, int n, int nPanels) {
  const int kl = k & 31;
  const int panel = (k >> 5) * nPanels + (n >> 4);
  const int lane = (n & 15) + ((kl & 8) << 1);
  const int j = ((kl >> 4) << 3) | (kl & 7);
  return panel * 512 + lane * 16 + j;
}
// Fragment loads: one 32-byte contiguous chunk per lane (2x ds_load_b128).
DEV v16h ldFragA(const _Float16* S, int kPanels, int mb, int kb, int lane) {
  return *(const v16h*)(S + (mb * kPanels + kb) * 512 + lane * 16);
}
DEV v16h ldFragB(const _Float16* S, int nPanels, int kb, int nb, int lane) {
  return *(const v16h*)(S + (kb * nPanels + nb) * 512 + lane * 16);
}
// Store a 16x16 f32 D tile into a swizzled A-type buffer (as f16).
DEV void stD_A(_Float16* S, int kPanels, int mb, int col0, int lane, v8f acc) {
  const int n = col0 + (lane & 15);
  const int mBase = (mb << 4) + ((lane & 16) ? 8 : 0);
#pragma unroll
  for (int v = 0; v < 8; ++v) S[idxA(mBase + v, n, kPanels)] = (_Float16)acc[v];
}

DEV unsigned int pk2(float a, float b) {  // pack 2 floats -> 2 f16 in a dword
  union { _Float16 h[2]; unsigned int u; } x;
  x.h[0] = (_Float16)a; x.h[1] = (_Float16)b;
  return x.u;
}
DEV float silu(float x) { return x / (1.0f + __expf(-x)); }

// ---- fused kernel -------------------------------------------------------
__global__ __launch_bounds__(TPB) void pe_crystal_fused(
    const float* __restrict__ pos,    // (B,N,3)
    const float* __restrict__ recip,  // (B,3,3)
    const int*   __restrict__ sgrp,   // (B,1)
    const float* __restrict__ abc,    // (K,3)
    const float* __restrict__ graphs, // (NSG,K,K)
    const float* __restrict__ wre,    // (NSG,K,K)
    const float* __restrict__ wim,    // (NSG,K,K)
    const float* __restrict__ bre,    // (NSG,K)
    const float* __restrict__ bim,    // (NSG,K)
    const float* __restrict__ W2,     // (E,E)
    const float* __restrict__ b2,     // (E)
    const float* __restrict__ gam,    // (E)
    const float* __restrict__ bet,    // (E)
    float* __restrict__ out)          // (B,N,E)
{
  __shared__ __align__(1024) unsigned char smem[SMEM_BYTES];

  _Float16* cosA = (_Float16*)(smem + OFF_COSA);
  _Float16* sinA = (_Float16*)(smem + OFF_SINA);
  _Float16* xA   = (_Float16*)(smem + OFF_XA);
  _Float16* adjB = (_Float16*)(smem + OFF_ADJ);
  _Float16* wRl  = (_Float16*)(smem + OFF_WR);
  _Float16* wIl  = (_Float16*)(smem + OFF_WI);
  _Float16* wrB  = (_Float16*)(smem + OFF_WRB);
  _Float16* wiB  = (_Float16*)(smem + OFF_WIB);
  _Float16* w2B  = (_Float16*)(smem + OFF_W2);
  float* xO    = (float*)(smem + OFF_XOUT);
  float* tabcL = (float*)(smem + OFF_TABC);
  float* posT  = (float*)(smem + OFF_POST);
  float* brL   = (float*)(smem + OFF_BR);
  float* biL   = (float*)(smem + OFF_BI);
  float* b2L   = (float*)(smem + OFF_B2);
  float* gamL  = (float*)(smem + OFF_GAM);
  float* betL  = (float*)(smem + OFF_BET);
  float* muL   = (float*)(smem + OFF_MU);
  float* rsL   = (float*)(smem + OFF_RS);

  const int b    = (int)blockIdx.x >> 4;          // N_/ROWS == 16 tiles/batch
  const int n0   = ((int)blockIdx.x & 15) * ROWS;
  const int tid  = (int)threadIdx.x;
  const int lane = tid & 31;
  const int wave = tid >> 5;
  const int sg   = sgrp[b] - 1;
  const size_t kk2 = (size_t)K_ * K_;

#if HAVE_TDM
  // ---- TDM: async-load W2 (64 KB f32) into LDS staging; overlaps
  //      stages 0-2. Wave 0 only (TDM ignores EXEC; avoid 8 dup DMAs).
  float* w2stg = (float*)(smem + OFF_W2STG);
  if (wave == 0) {
    const unsigned long long ga = (unsigned long long)(uintptr_t)W2;
    const unsigned int ldsOff = (unsigned int)(uintptr_t)w2stg;  // low 32b = LDS offset
    // D# group0: count=1 | lds_addr | global_addr[56:0] | type=2
    v4ui g0 = { 1u, ldsOff, (unsigned int)ga,
                ((unsigned int)(ga >> 32) & 0x01FFFFFFu) | 0x80000000u };
    // D# group1: data_size=4B; tensor 128x128, tile 128x128, stride0=128
    v8i g1 = { (int)(2u << 16),        // [17:16] data_size=2 (4 bytes)
               (int)(128u << 16),      // tensor_dim0[15:0] in [31:16]
               (int)(128u << 16),      // tensor_dim0[31:16]=0 | tensor_dim1[15:0]
               (int)(128u << 16),      // tensor_dim1[31:16]=0 | tile_dim0=128
               (int)128,               // tile_dim1=128 | tile_dim2=0
               (int)128,               // tensor_dim0_stride[31:0]=128
               0, 0 };
    v4i z4 = { 0, 0, 0, 0 };
#if defined(__clang_major__) && (__clang_major__ >= 23)
    v8i z8 = { 0, 0, 0, 0, 0, 0, 0, 0 };
    __builtin_amdgcn_tensor_load_to_lds(g0, g1, z4, z4, z8, 0);
#else
    __builtin_amdgcn_tensor_load_to_lds(g0, g1, z4, z4, 0);
#endif
  }
#endif

  // ---- P0: cooperative loads (small vectors + gathered sg matrices) -
  if (tid < K_) {
    brL[tid] = bre[(size_t)sg * K_ + tid];
    biL[tid] = bim[(size_t)sg * K_ + tid];
  }
  if (tid < E_) {
    b2L[tid]  = b2[tid];
    gamL[tid] = gam[tid];
    betL[tid] = bet[tid];
  }
  if (tid < 3 * ROWS) {  // 192: atom positions tile + t_abc (K x 3)
    const int q = tid / 3, c = tid % 3;
    posT[tid] = pos[((size_t)b * N_ + n0 + q) * 3 + c];
    float s = 0.0f;  // t_abc[q][c] = sum_d abc[q][d] * recip[b][c][d]
#pragma unroll
    for (int d = 0; d < 3; ++d)
      s += abc[q * 3 + d] * recip[(size_t)b * 9 + c * 3 + d];
    tabcL[tid] = s;
  }
  // adj / w_real / w_imag -> swizzled f16 B-operands (k-pairs -> b32 stores)
  for (int p = tid; p < (K_ / 2) * K_; p += TPB) {
    const int n = p & 63;
    const int k = (p >> 6) << 1;                 // even k; rows k and k+1
    const size_t g = (size_t)sg * kk2 + (size_t)k * K_ + n;
    const int o = idxB(k, n, 4);                 // j even -> dword aligned
    *(unsigned int*)(adjB + o) = pk2(graphs[g], graphs[g + K_]);
    *(unsigned int*)(wrB + o)  = pk2(wre[g],    wre[g + K_]);
    *(unsigned int*)(wiB + o)  = pk2(wim[g],    wim[g + K_]);
  }
  __syncthreads();

  // ---- stage 1a: phase -> cos/sin (fp32 math, swizzled f16 A) -------
  for (int p = tid; p < ROWS * (K_ / 2); p += TPB) {
    const int r = p >> 5;
    const int k = (p & 31) << 1;                 // even k: k, k+1
    const float px = posT[r * 3 + 0], py = posT[r * 3 + 1], pz = posT[r * 3 + 2];
    const float ph0 = TWO_PI * (px * tabcL[k * 3 + 0] + py * tabcL[k * 3 + 1] +
                                pz * tabcL[k * 3 + 2]);
    const float ph1 = TWO_PI * (px * tabcL[k * 3 + 3] + py * tabcL[k * 3 + 4] +
                                pz * tabcL[k * 3 + 5]);
    float s0, c0, s1, c1;
    __sincosf(ph0, &s0, &c0);
    __sincosf(ph1, &s1, &c1);
    const int o = idxA(r, k, 2);
    *(unsigned int*)(cosA + o) = pk2(c0, c1);
    *(unsigned int*)(sinA + o) = pk2(s0, s1);
  }
  __syncthreads();

  // ---- stage 1b: weighted = enc @ adj (2 GEMMs, 64x64x64) -----------
#pragma unroll
  for (int it = 0; it < 4; ++it) {
    const int id = wave + 8 * it;                // 32 tiles over 8 waves
    const int g = id >> 4, t = id & 15;
    const int mb = t >> 2, nb = t & 3;
    const _Float16* A = g ? sinA : cosA;
    _Float16* D = g ? wIl : wRl;
    v8f acc = {0.f, 0.f, 0.f, 0.f, 0.f, 0.f, 0.f, 0.f};
    acc = wmma_f16(ldFragA(A, 2, mb, 0, lane), ldFragB(adjB, 4, 0, nb, lane), acc);
    acc = wmma_f16(ldFragA(A, 2, mb, 1, lane), ldFragB(adjB, 4, 1, nb, lane), acc);
    stD_A(D, 2, mb, nb << 4, lane, acc);
  }
  __syncthreads();

  // ---- stage 2: complex GEMM + bias + SiLU -> x (64x128 f16 A) ------
#pragma unroll
  for (int it = 0; it < 4; ++it) {
    const int id = wave + 8 * it;
    const int g = id >> 4, t = id & 15;
    const int mb = t >> 2, nb = t & 3;
    v8f acc = {0.f, 0.f, 0.f, 0.f, 0.f, 0.f, 0.f, 0.f};
    if (g == 0) {  // Re = wR@wr - wI@wi
      v8f acc2 = {0.f, 0.f, 0.f, 0.f, 0.f, 0.f, 0.f, 0.f};
      acc  = wmma_f16(ldFragA(wRl, 2, mb, 0, lane), ldFragB(wrB, 4, 0, nb, lane), acc);
      acc  = wmma_f16(ldFragA(wRl, 2, mb, 1, lane), ldFragB(wrB, 4, 1, nb, lane), acc);
      acc2 = wmma_f16(ldFragA(wIl, 2, mb, 0, lane), ldFragB(wiB, 4, 0, nb, lane), acc2);
      acc2 = wmma_f16(ldFragA(wIl, 2, mb, 1, lane), ldFragB(wiB, 4, 1, nb, lane), acc2);
      acc = acc - acc2;
    } else {       // Im = wR@wi + wI@wr (chained accumulate)
      acc = wmma_f16(ldFragA(wRl, 2, mb, 0, lane), ldFragB(wiB, 4, 0, nb, lane), acc);
      acc = wmma_f16(ldFragA(wRl, 2, mb, 1, lane), ldFragB(wiB, 4, 1, nb, lane), acc);
      acc = wmma_f16(ldFragA(wIl, 2, mb, 0, lane), ldFragB(wrB, 4, 0, nb, lane), acc);
      acc = wmma_f16(ldFragA(wIl, 2, mb, 1, lane), ldFragB(wrB, 4, 1, nb, lane), acc);
    }
    const int f = (nb << 4) + (lane & 15);       // feature inside K_
    const int n = (g << 6) + f;                  // column inside E_
    const int mBase = (mb << 4) + ((lane & 16) ? 8 : 0);
    const float bb = (g ? biL : brL)[f];
#pragma unroll
    for (int v = 0; v < 8; ++v)
      xA[idxA(mBase + v, n, 4)] = (_Float16)silu(acc[v] + bb);
  }
  __syncthreads();

  // ---- W2^T -> swizzled f16 B (128x128) over freed LDS --------------
#if HAVE_TDM
  if (wave == 0) __builtin_amdgcn_s_wait_tensorcnt(0);
  __syncthreads();                                // TDM data visible to all
  const float* w2src = w2stg;                     // row-major W2 in LDS
#else
  const float* w2src = W2;                        // fallback: straight global
#endif
  for (int p = tid; p < E_ * (E_ / 2); p += TPB) {
    const int e = p >> 6;
    const int f = (p & 63) << 1;                 // even f: f, f+1 contiguous
    const float* s = w2src + (size_t)e * E_ + f;
    *(unsigned int*)(w2B + idxB(f, e, 8)) = pk2(s[0], s[1]);
  }
  __syncthreads();

  // ---- stage 3: y = x @ W2^T + b2 (64x128x128) ----------------------
#pragma unroll
  for (int it = 0; it < 4; ++it) {
    const int id = wave + 8 * it;                // 4x8 tiles over 8 waves
    const int mb = id >> 3, eb = id & 7;
    v8f acc = {0.f, 0.f, 0.f, 0.f, 0.f, 0.f, 0.f, 0.f};
#pragma unroll
    for (int kb = 0; kb < 4; ++kb)
      acc = wmma_f16(ldFragA(xA, 4, mb, kb, lane), ldFragB(w2B, 8, kb, eb, lane), acc);
    const int e = (eb << 4) + (lane & 15);
    const int mBase = (mb << 4) + ((lane & 16) ? 8 : 0);
    const float bb = b2L[e];
    float* yp = xO + (size_t)mBase * E_ + e;
#pragma unroll
    for (int v = 0; v < 8; ++v) yp[v * E_] = acc[v] + bb;
  }
  __syncthreads();

  // ---- stage 4: LayerNorm(128) + single HBM write -------------------
  if (tid < ROWS) {
    float s = 0.f, s2 = 0.f;
#pragma unroll 8
    for (int e = 0; e < E_; ++e) {
      const float x = xO[tid * E_ + e];
      s += x; s2 += x * x;
    }
    const float mu = s * (1.0f / E_);
    muL[tid] = mu;
    rsL[tid] = rsqrtf(s2 * (1.0f / E_) - mu * mu + LN_EPS);
  }
  __syncthreads();
  for (int i = tid; i < ROWS * E_; i += TPB) {
    const int r = i >> 7, e = i & 127;
    out[((size_t)b * N_ + n0 + r) * E_ + e] =
        (xO[i] - muL[r]) * rsL[r] * gamL[e] + betL[e];
  }
}

// ---- host entry ---------------------------------------------------------
extern "C" void kernel_launch(void* const* d_in, const int* in_sizes, int n_in,
                              void* d_out, int out_size, void* d_ws, size_t ws_size,
                              hipStream_t stream) {
  (void)in_sizes; (void)n_in; (void)out_size; (void)d_ws; (void)ws_size;
  const float* pos    = (const float*)d_in[0];
  const float* recip  = (const float*)d_in[1];
  const int*   sgrp   = (const int*)  d_in[2];
  const float* abc    = (const float*)d_in[3];
  const float* graphs = (const float*)d_in[4];
  const float* wre    = (const float*)d_in[5];
  const float* wim    = (const float*)d_in[6];
  const float* bre    = (const float*)d_in[7];
  const float* bim    = (const float*)d_in[8];
  const float* W2     = (const float*)d_in[9];
  const float* b2     = (const float*)d_in[10];
  const float* gam    = (const float*)d_in[11];
  const float* bet    = (const float*)d_in[12];
  float* out = (float*)d_out;

  dim3 grid(B_ * (N_ / ROWS));   // 4096 workgroups
  dim3 block(TPB);               // 8 wave32
  pe_crystal_fused<<<grid, block, 0, stream>>>(
      pos, recip, sgrp, abc, graphs, wre, wim, bre, bim, W2, b2, gam, bet, out);
}
